// PointTransformerLayer_24988119728694
// MI455X (gfx1250) — matrix operand order repaired and checked by reading, not compiled, not run on hardware
//
#include <hip/hip_runtime.h>
#include <hip/hip_bf16.h>
#include <stdint.h>

#define BB   2
#define NN   8192
#define CC   128
#define KK   16
#define EPSV 1e-5f
#define TAS  136   // bf16 A-tile row stride (elements) — padded vs 64-bank LDS
#define KVS  132   // f32 K/V-tile row stride (elements)

typedef __attribute__((ext_vector_type(16))) __bf16       v16bf;
typedef __attribute__((ext_vector_type(8)))  float        v8f;
typedef __attribute__((ext_vector_type(8)))  unsigned int v8u;

// ---------- helpers ----------

__device__ __forceinline__ unsigned short f32_bf16(float f) {
  unsigned u = __builtin_bit_cast(unsigned, f);
  u += 0x7FFFu + ((u >> 16) & 1u);          // round-to-nearest-even
  return (unsigned short)(u >> 16);
}

// A-fragment (bf16 16x32) from row-major LDS tile, contiguous k order (same as b_frag):
// lane = hl*16+ml holds row M=ml; VGPR v holds K = ks*32 + hl*16 + 2*v, +1
__device__ __forceinline__ v16bf a_frag(const unsigned short* t, int ml, int hl, int ks) {
  v8u u;
#pragma unroll
  for (int v = 0; v < 8; ++v) {
    int k = ks * 32 + hl * 16 + 2 * v;
    u[v] = *(const unsigned int*)(t + ml * TAS + k);   // 2 packed bf16
  }
  return __builtin_bit_cast(v16bf, u);
}

// B-fragment (bf16 32x16) straight from global bf16 weight, torch [out,in] row-major.
// B[k][n] = W[n][k]; lane n = nt*16+(lane&15) reads contiguous 16 bf16 (32B) of row n.
__device__ __forceinline__ v16bf b_frag(const unsigned short* __restrict__ w,
                                        int nt, int ks, int lane) {
  int nrow = nt * 16 + (lane & 15);
  int k0   = ks * 32 + (lane >> 4) * 16;
  return *(const v16bf*)(w + nrow * CC + k0);
}

// ---------- prep: weights -> bf16, fold BN affine ----------

__global__ void __launch_bounds__(256)
prep_w_kernel(const float* __restrict__ Wq,  const float* __restrict__ Wk,
              const float* __restrict__ Wv,  const float* __restrict__ pW2,
              const float* __restrict__ aW1, const float* __restrict__ aW2,
              const float* __restrict__ Wout,
              unsigned short* o0, unsigned short* o1, unsigned short* o2,
              unsigned short* o3, unsigned short* o4, unsigned short* o5,
              unsigned short* o6) {
  int e = blockIdx.x * 256 + threadIdx.x;   // 0..16383
  o0[e] = f32_bf16(Wq[e]);  o1[e] = f32_bf16(Wk[e]);  o2[e] = f32_bf16(Wv[e]);
  o3[e] = f32_bf16(pW2[e]); o4[e] = f32_bf16(aW1[e]); o5[e] = f32_bf16(aW2[e]);
  o6[e] = f32_bf16(Wout[e]);
}

// bn(x + b) = x*scale + shift, scale = g*rsqrt(var+eps), shift = (b-mean)*scale + beta
__global__ void prep_bn_kernel(const float* pb1, const float* pg, const float* pbeta,
                               const float* pmean, const float* pvar,
                               const float* ab1, const float* ag, const float* abeta,
                               const float* amean, const float* avar,
                               float* pscaleE, float* pshiftE,
                               float* ascaleE, float* ashiftE) {
  int c = threadIdx.x;                       // 128 threads
  float ps = pg[c] * rsqrtf(pvar[c] + EPSV);
  pscaleE[c] = ps;
  pshiftE[c] = (pb1[c] - pmean[c]) * ps + pbeta[c];
  float as = ag[c] * rsqrtf(avar[c] + EPSV);
  ascaleE[c] = as;
  ashiftE[c] = (ab1[c] - amean[c]) * as + abeta[c];
}

// ---------- fused KNN: stream all points, register top-16 ----------

__global__ void __launch_bounds__(256)
knn_kernel(const float* __restrict__ xyz, int* __restrict__ knn_idx) {
  __shared__ float sx[256], sy[256], sz[256];
  const int gid = blockIdx.x * 256 + threadIdx.x;  // 0..16383 (blocks batch-aligned)
  const int b = gid >> 13;
  const int n = gid & (NN - 1);
  const float* base = xyz + (size_t)b * NN * 3;
  const float qx = base[n * 3 + 0], qy = base[n * 3 + 1], qz = base[n * 3 + 2];

  float bd[KK]; int bi[KK];
#pragma unroll
  for (int i = 0; i < KK; ++i) { bd[i] = 3.4e38f; bi[i] = 0; }

  for (int c0 = 0; c0 < NN; c0 += 256) {
    __syncthreads();
    int p = c0 + threadIdx.x;
    sx[threadIdx.x] = base[p * 3 + 0];
    sy[threadIdx.x] = base[p * 3 + 1];
    sz[threadIdx.x] = base[p * 3 + 2];
    if (c0 + 256 < NN)
      __builtin_prefetch(base + (size_t)(c0 + 256 + threadIdx.x) * 3, 0, 0);
    __syncthreads();
    for (int j = 0; j < 256; ++j) {
      float dx = sx[j] - qx, dy = sy[j] - qy, dz = sz[j] - qz;
      float d = dx * dx + dy * dy + dz * dz;
      if (d < bd[KK - 1]) {                  // branch-free predicated insert
        int cand = c0 + j;
#pragma unroll
        for (int t = KK - 1; t > 0; --t) {
          if (bd[t] > d) {
            if (bd[t - 1] > d) { bd[t] = bd[t - 1]; bi[t] = bi[t - 1]; }
            else              { bd[t] = d;          bi[t] = cand;      }
          }
        }
        if (bd[0] > d) { bd[0] = d; bi[0] = cand; }
      }
    }
  }
  int* o = knn_idx + (size_t)gid * KK;
#pragma unroll
  for (int i = 0; i < KK; ++i) o[i] = bi[i];
}

// ---------- QKV projection: gather-commuted, per-point [16384,128]x[128,128]x3 ----------

__global__ void __launch_bounds__(128)
qkv_kernel(const float* __restrict__ feat,
           const unsigned short* __restrict__ wq,
           const unsigned short* __restrict__ wk,
           const unsigned short* __restrict__ wv,
           float* __restrict__ qb, float* __restrict__ kb, float* __restrict__ vb) {
  __shared__ unsigned short tA[4][16 * TAS];
  const int wave = threadIdx.x >> 5, lane = threadIdx.x & 31;
  const int ml = lane & 15, hl = lane >> 4;
  const int row0 = blockIdx.x * 64 + wave * 16;
  unsigned short* myA = &tA[wave][0];

#pragma unroll
  for (int m = 0; m < 16; ++m) {             // stage A-tile once (bf16), reused for 3 weights
    float4 f = ((const float4*)(feat + (size_t)(row0 + m) * CC))[lane];
    myA[m * TAS + lane * 4 + 0] = f32_bf16(f.x);
    myA[m * TAS + lane * 4 + 1] = f32_bf16(f.y);
    myA[m * TAS + lane * 4 + 2] = f32_bf16(f.z);
    myA[m * TAS + lane * 4 + 3] = f32_bf16(f.w);
  }

  const unsigned short* wlist[3] = {wq, wk, wv};
  float* olist[3] = {qb, kb, vb};
#pragma unroll
  for (int s = 0; s < 3; ++s) {
    const unsigned short* w = wlist[s];
    float* o = olist[s];
#pragma unroll
    for (int nt = 0; nt < 8; ++nt) {
      v8f acc = {};
#pragma unroll
      for (int ks = 0; ks < 4; ++ks)
        acc = __builtin_amdgcn_wmma_f32_16x16x32_bf16(
            false, a_frag(myA, ml, hl, ks), false, b_frag(w, nt, ks, lane),
            (short)0, acc, false, false);
      int c = nt * 16 + ml;
#pragma unroll
      for (int v = 0; v < 8; ++v)
        o[(size_t)(row0 + hl * 8 + v) * CC + c] = acc[v];
    }
  }
}

// ---------- fused attention: one wave == one query == one 16-row WMMA tile ----------

__global__ void __launch_bounds__(128)
fused_attn_kernel(const float* __restrict__ xyz, const int* __restrict__ knn_idx,
                  const float* __restrict__ qbuf, const float* __restrict__ kbuf,
                  const float* __restrict__ vbuf,
                  const float* __restrict__ pW1,
                  const float* __restrict__ pscaleE, const float* __restrict__ pshiftE,
                  const float* __restrict__ pb2,
                  const float* __restrict__ ascaleE, const float* __restrict__ ashiftE,
                  const float* __restrict__ ab2,
                  const unsigned short* __restrict__ wpw2,
                  const unsigned short* __restrict__ waw1,
                  const unsigned short* __restrict__ waw2,
                  float* __restrict__ obuf) {
  __shared__ unsigned short tA[4][16 * TAS];   // bf16 A-tile, reused for 3 GEMM stages
  __shared__ float tKV[4][16 * KVS];           // fp32 tile: K rows, later V rows
  __shared__ float tQ[4][CC];

  const int wave = threadIdx.x >> 5, lane = threadIdx.x & 31;
  const int ml = lane & 15, hl = lane >> 4;
  const int qid = blockIdx.x * 4 + wave;       // 0..16383
  const int b = qid / NN, n = qid % NN;
  unsigned short* myA = &tA[wave][0];
  float* myKV = &tKV[wave][0];
  float* myQ  = &tQ[wave][0];

  // neighbor id per half-lane (lanes 16..31 duplicate 0..15)
  const int nb = knn_idx[(size_t)qid * KK + ml];
  const float qx = xyz[((size_t)b * NN + n) * 3 + 0];
  const float qy = xyz[((size_t)b * NN + n) * 3 + 1];
  const float qz = xyz[((size_t)b * NN + n) * 3 + 2];
  const float rx = xyz[((size_t)b * NN + nb) * 3 + 0] - qx;
  const float ry = xyz[((size_t)b * NN + nb) * 3 + 1] - qy;
  const float rz = xyz[((size_t)b * NN + nb) * 3 + 2] - qz;

  { // stage q row (128 f32)
    float4 qv = ((const float4*)(qbuf + (size_t)qid * CC))[lane];
    myQ[lane * 4 + 0] = qv.x; myQ[lane * 4 + 1] = qv.y;
    myQ[lane * 4 + 2] = qv.z; myQ[lane * 4 + 3] = qv.w;
  }
#pragma unroll
  for (int m = 0; m < KK; ++m) {               // gather K rows (coalesced float4)
    int nm = __shfl(nb, m, 32);
    float4 kv = ((const float4*)(kbuf + ((size_t)b * NN + nm) * CC))[lane];
    myKV[m * KVS + lane * 4 + 0] = kv.x; myKV[m * KVS + lane * 4 + 1] = kv.y;
    myKV[m * KVS + lane * 4 + 2] = kv.z; myKV[m * KVS + lane * 4 + 3] = kv.w;
  }

  // per-lane channel params
  float w0[4], w1[4], w2[4], psc[4], psh[4];
#pragma unroll
  for (int cc = 0; cc < 4; ++cc) {
    int c = lane + 32 * cc;
    w0[cc] = pW1[c * 3 + 0]; w1[cc] = pW1[c * 3 + 1]; w2[cc] = pW1[c * 3 + 2];
    psc[cc] = pscaleE[c];    psh[cc] = pshiftE[c];
  }
  float pb2c[8], asc[8], ash[8], ab2c[8];
#pragma unroll
  for (int nt = 0; nt < 8; ++nt) {
    int c = nt * 16 + ml;
    pb2c[nt] = pb2[c]; asc[nt] = ascaleE[c]; ash[nt] = ashiftE[c]; ab2c[nt] = ab2[c];
  }

  // h1 = relu(bn(rel @ pW1 + pb1))  -> tA (bf16)
#pragma unroll
  for (int m = 0; m < KK; ++m) {
    float rmx = __shfl(rx, m, 32), rmy = __shfl(ry, m, 32), rmz = __shfl(rz, m, 32);
#pragma unroll
    for (int cc = 0; cc < 4; ++cc) {
      float x = rmx * w0[cc] + rmy * w1[cc] + rmz * w2[cc];
      float h = fmaxf(x * psc[cc] + psh[cc], 0.0f);
      myA[m * TAS + lane + 32 * cc] = f32_bf16(h);
    }
  }

  // pos = h1 @ pW2^T + pb2   (WMMA, fp32 accum kept live in registers)
  v8f posAcc[8];
#pragma unroll
  for (int nt = 0; nt < 8; ++nt) {
    v8f acc = {};
#pragma unroll
    for (int ks = 0; ks < 4; ++ks)
      acc = __builtin_amdgcn_wmma_f32_16x16x32_bf16(
          false, a_frag(myA, ml, hl, ks), false, b_frag(wpw2, nt, ks, lane),
          (short)0, acc, false, false);
#pragma unroll
    for (int v = 0; v < 8; ++v) acc[v] += pb2c[nt];
    posAcc[nt] = acc;
  }

  // attn_pre = q - k + pos  -> tA (overwrite; C/D layout -> row-major LDS)
#pragma unroll
  for (int nt = 0; nt < 8; ++nt) {
    int c = nt * 16 + ml;
    float qc = myQ[c];
#pragma unroll
    for (int v = 0; v < 8; ++v) {
      int m = hl * 8 + v;
      myA[m * TAS + c] = f32_bf16(qc - myKV[m * KVS + c] + posAcc[nt][v]);
    }
  }

  // K reads done — overwrite tKV with gathered V rows
#pragma unroll
  for (int m = 0; m < KK; ++m) {
    int nm = __shfl(nb, m, 32);
    float4 vv = ((const float4*)(vbuf + ((size_t)b * NN + nm) * CC))[lane];
    myKV[m * KVS + lane * 4 + 0] = vv.x; myKV[m * KVS + lane * 4 + 1] = vv.y;
    myKV[m * KVS + lane * 4 + 2] = vv.z; myKV[m * KVS + lane * 4 + 3] = vv.w;
  }

  // attn MLP stage 1: @aW1^T, BN+ReLU -> tA
  v8f g[8];
#pragma unroll
  for (int nt = 0; nt < 8; ++nt) {
    v8f acc = {};
#pragma unroll
    for (int ks = 0; ks < 4; ++ks)
      acc = __builtin_amdgcn_wmma_f32_16x16x32_bf16(
          false, a_frag(myA, ml, hl, ks), false, b_frag(waw1, nt, ks, lane),
          (short)0, acc, false, false);
    g[nt] = acc;
  }
#pragma unroll
  for (int nt = 0; nt < 8; ++nt) {
    int c = nt * 16 + ml;
#pragma unroll
    for (int v = 0; v < 8; ++v) {
      int m = hl * 8 + v;
      myA[m * TAS + c] = f32_bf16(fmaxf(g[nt][v] * asc[nt] + ash[nt], 0.0f));
    }
  }

  // attn MLP stage 2: @aW2^T + ab2
#pragma unroll
  for (int nt = 0; nt < 8; ++nt) {
    v8f acc = {};
#pragma unroll
    for (int ks = 0; ks < 4; ++ks)
      acc = __builtin_amdgcn_wmma_f32_16x16x32_bf16(
          false, a_frag(myA, ml, hl, ks), false, b_frag(waw2, nt, ks, lane),
          (short)0, acc, false, false);
#pragma unroll
    for (int v = 0; v < 8; ++v) acc[v] += ab2c[nt];
    g[nt] = acc;
  }

  // per-channel softmax over K=16 neighbors (8 local rows + partner half-wave)
#pragma unroll
  for (int nt = 0; nt < 8; ++nt) {
    v8f t = g[nt];
    float mx = t[0];
#pragma unroll
    for (int v = 1; v < 8; ++v) mx = fmaxf(mx, t[v]);
    mx = fmaxf(mx, __shfl_xor(mx, 16, 32));
    float s = 0.0f;
#pragma unroll
    for (int v = 0; v < 8; ++v) { float e = __expf(t[v] - mx); t[v] = e; s += e; }
    s += __shfl_xor(s, 16, 32);
    float inv = 1.0f / s;
#pragma unroll
    for (int v = 0; v < 8; ++v) t[v] *= inv;
    g[nt] = t;
  }

  // out[c] = sum_k attn[k,c] * (v[k,c] + pos[k,c])
#pragma unroll
  for (int nt = 0; nt < 8; ++nt) {
    int c = nt * 16 + ml;
    float o = 0.0f;
#pragma unroll
    for (int v = 0; v < 8; ++v) {
      int m = hl * 8 + v;
      o += g[nt][v] * (myKV[m * KVS + c] + posAcc[nt][v]);
    }
    o += __shfl_xor(o, 16, 32);
    if (lane < 16) obuf[(size_t)qid * CC + c] = o;
  }
}

// ---------- final Wout GEMM: [16384,128] x [128,128] ----------

__global__ void __launch_bounds__(128)
rowgemm_kernel(const float* __restrict__ in, const unsigned short* __restrict__ w,
               float* __restrict__ out) {
  __shared__ unsigned short tA[4][16 * TAS];
  const int wave = threadIdx.x >> 5, lane = threadIdx.x & 31;
  const int ml = lane & 15, hl = lane >> 4;
  const int row0 = blockIdx.x * 64 + wave * 16;
  unsigned short* myA = &tA[wave][0];
#pragma unroll
  for (int m = 0; m < 16; ++m) {
    float4 f = ((const float4*)(in + (size_t)(row0 + m) * CC))[lane];
    myA[m * TAS + lane * 4 + 0] = f32_bf16(f.x);
    myA[m * TAS + lane * 4 + 1] = f32_bf16(f.y);
    myA[m * TAS + lane * 4 + 2] = f32_bf16(f.z);
    myA[m * TAS + lane * 4 + 3] = f32_bf16(f.w);
  }
#pragma unroll
  for (int nt = 0; nt < 8; ++nt) {
    v8f acc = {};
#pragma unroll
    for (int ks = 0; ks < 4; ++ks)
      acc = __builtin_amdgcn_wmma_f32_16x16x32_bf16(
          false, a_frag(myA, ml, hl, ks), false, b_frag(w, nt, ks, lane),
          (short)0, acc, false, false);
    int c = nt * 16 + ml;
#pragma unroll
    for (int v = 0; v < 8; ++v)
      out[(size_t)(row0 + hl * 8 + v) * CC + c] = acc[v];
  }
}

// ---------- launch ----------

extern "C" void kernel_launch(void* const* d_in, const int* in_sizes, int n_in,
                              void* d_out, int out_size, void* d_ws, size_t ws_size,
                              hipStream_t stream) {
  (void)in_sizes; (void)n_in; (void)out_size; (void)ws_size;
  const float* xyz   = (const float*)d_in[0];
  const float* feat  = (const float*)d_in[1];
  const float* Wq    = (const float*)d_in[2];
  const float* Wk    = (const float*)d_in[3];
  const float* Wv    = (const float*)d_in[4];
  const float* pW1   = (const float*)d_in[5];
  const float* pb1   = (const float*)d_in[6];
  const float* pg    = (const float*)d_in[7];
  const float* pbeta = (const float*)d_in[8];
  const float* pmean = (const float*)d_in[9];
  const float* pvar  = (const float*)d_in[10];
  const float* pW2   = (const float*)d_in[11];
  const float* pb2   = (const float*)d_in[12];
  const float* aW1   = (const float*)d_in[13];
  const float* ab1   = (const float*)d_in[14];
  const float* ag    = (const float*)d_in[15];
  const float* abeta = (const float*)d_in[16];
  const float* amean = (const float*)d_in[17];
  const float* avar  = (const float*)d_in[18];
  const float* aW2   = (const float*)d_in[19];
  const float* ab2   = (const float*)d_in[20];
  const float* Wout  = (const float*)d_in[21];

  char* ws = (char*)d_ws;
  const size_t FB = (size_t)BB * NN * CC * sizeof(float);   // 8 MB per feature buffer
  int*   knn_idx = (int*)(ws);                              // 1 MB
  float* qb = (float*)(ws + (1u << 20));
  float* kb = (float*)(ws + (1u << 20) + 1 * FB);
  float* vb = (float*)(ws + (1u << 20) + 2 * FB);
  float* ob = (float*)(ws + (1u << 20) + 3 * FB);
  unsigned short* wq16   = (unsigned short*)(ws + (1u << 20) + 4 * FB);
  unsigned short* wk16   = wq16 + CC * CC;
  unsigned short* wv16   = wk16 + CC * CC;
  unsigned short* pw216  = wv16 + CC * CC;
  unsigned short* aw116  = pw216 + CC * CC;
  unsigned short* aw216  = aw116 + CC * CC;
  unsigned short* wout16 = aw216 + CC * CC;
  float* pscaleE = (float*)(wout16 + CC * CC);
  float* pshiftE = pscaleE + CC;
  float* ascaleE = pshiftE + CC;
  float* ashiftE = ascaleE + CC;

  prep_w_kernel<<<64, 256, 0, stream>>>(Wq, Wk, Wv, pW2, aW1, aW2, Wout,
                                        wq16, wk16, wv16, pw216, aw116, aw216, wout16);
  prep_bn_kernel<<<1, 128, 0, stream>>>(pb1, pg, pbeta, pmean, pvar,
                                        ab1, ag, abeta, amean, avar,
                                        pscaleE, pshiftE, ascaleE, ashiftE);
  knn_kernel<<<(BB * NN) / 256, 256, 0, stream>>>(xyz, knn_idx);
  qkv_kernel<<<(BB * NN) / 64, 128, 0, stream>>>(feat, wq16, wk16, wv16, qb, kb, vb);
  fused_attn_kernel<<<(BB * NN) / 4, 128, 0, stream>>>(
      xyz, knn_idx, qb, kb, vb, pW1, pscaleE, pshiftE, pb2,
      ascaleE, ashiftE, ab2, pw216, aw116, aw216, ob);
  rowgemm_kernel<<<(BB * NN) / 64, 128, 0, stream>>>(ob, wout16, (float*)d_out);
}
